// ScaledDotProductAttention_49220325212479
// MI455X (gfx1250) — matrix operand order repaired and checked by reading, not compiled, not run on hardware
//
#include <hip/hip_runtime.h>

// Problem constants from the reference
#define BB 4
#define HH 16
#define SQ 1024
#define SK 1024
#define HD 64
#define LN_EPS 1e-5f
#define NEG_INF -1e9f

typedef __attribute__((ext_vector_type(16))) _Float16 v16h;
typedef __attribute__((ext_vector_type(8)))  float    v8f;
typedef __attribute__((ext_vector_type(4)))  float    v4f;
typedef __attribute__((ext_vector_type(4)))  int      v4i;

__device__ __forceinline__ float wave_sum(float v) {
  #pragma unroll
  for (int o = 16; o > 0; o >>= 1) v += __shfl_xor(v, o, 32);
  return v;
}
__device__ __forceinline__ float wave_max(float v) {
  #pragma unroll
  for (int o = 16; o > 0; o >>= 1) v = fmaxf(v, __shfl_xor(v, o, 32));
  return v;
}

// One workgroup = one (b, h, 16-row q-tile). 128 threads = 4 waves (wave32).
__global__ __launch_bounds__(128, 1)
void attn_ln_softmax_kernel(const float* __restrict__ Q,
                            const float* __restrict__ K,
                            const float* __restrict__ V,
                            const float* __restrict__ gamma,
                            const float* __restrict__ beta,
                            const int*   __restrict__ mask,
                            float* __restrict__ out,
                            float* __restrict__ attn)
{
  __shared__ float sc[16 * SK];                 // 64 KB score/prob tile, LDS-resident

  const int qt = blockIdx.x;                    // q-tile (0..63)
  const int h  = blockIdx.y;
  const int b  = blockIdx.z;
  const int bh = b * HH + h;
  const int q0 = qt * 16;

  const int tid  = threadIdx.x;
  const int w    = tid >> 5;                    // wave id 0..3
  const int lane = tid & 31;
  const int n    = lane & 15;                   // M (for A) / N (for B,C,D)
  const int hf   = lane >> 4;                   // lane half select

  // ---------------- Phase 1: A fragments (Q tile), shared by all N-tiles ----
  v16h a0, a1;                                  // K = 0..31 and K = 32..63
  {
    const float* qrow = Q + ((size_t)bh * SQ + q0 + n) * HD;
    #pragma unroll
    for (int v = 0; v < 8; ++v) {
      const int k0 = ((v >> 2) << 4) + ((v & 3) << 1) + (hf << 3);
      a0[2 * v]     = (_Float16)qrow[k0];
      a0[2 * v + 1] = (_Float16)qrow[k0 + 1];
      a1[2 * v]     = (_Float16)qrow[32 + k0];
      a1[2 * v + 1] = (_Float16)qrow[32 + k0 + 1];
    }
  }

  // ---------------- Phase 2: S = (Q K^T) / sqrt(64) into LDS ----------------
  #pragma unroll 2
  for (int i = 0; i < 16; ++i) {
    const int t = w * 16 + i;                   // N-tile (0..63)
    v16h b0f, b1f;
    const float* krow = K + ((size_t)bh * SK + t * 16 + n) * HD;
    #pragma unroll
    for (int v = 0; v < 8; ++v) {
      const int k0 = ((v >> 2) << 4) + ((v & 3) << 1) + (hf << 3);
      b0f[2 * v]     = (_Float16)krow[k0];
      b0f[2 * v + 1] = (_Float16)krow[k0 + 1];
      b1f[2 * v]     = (_Float16)krow[32 + k0];
      b1f[2 * v + 1] = (_Float16)krow[32 + k0 + 1];
    }
    v8f acc = {};
    acc = __builtin_amdgcn_wmma_f32_16x16x32_f16(false, a0, false, b0f,
                                                 (short)0, acc, false, false);
    acc = __builtin_amdgcn_wmma_f32_16x16x32_f16(false, a1, false, b1f,
                                                 (short)0, acc, false, false);
    #pragma unroll
    for (int r = 0; r < 8; ++r)
      sc[(r + 8 * hf) * SK + t * 16 + n] = acc[r] * 0.125f;  // 1/sqrt(64)
  }
  __syncthreads();

  // ---- Phase 3: per-row LayerNorm + mask + softmax (wave owns 4 rows) ------
  // All sweeps are 128-bit per lane: one wave iteration = 512B contiguous.
  for (int rr = 0; rr < 4; ++rr) {
    const int row = w * 4 + rr;
    float* srow = sc + row * SK;

    float s = 0.f, s2 = 0.f;
    #pragma unroll
    for (int it = 0; it < 8; ++it) {
      const int k = it * 128 + lane * 4;
      v4f x = *(const v4f*)(srow + k);
      #pragma unroll
      for (int j = 0; j < 4; ++j) { s += x[j]; s2 += x[j] * x[j]; }
    }
    s  = wave_sum(s);
    s2 = wave_sum(s2);
    const float mean = s * (1.0f / SK);
    const float var  = s2 * (1.0f / SK) - mean * mean;
    const float inv  = rsqrtf(var + LN_EPS);

    const int* mrow = mask + ((size_t)b * SQ + q0 + row) * SK;
    float mx = -INFINITY;
    #pragma unroll
    for (int it = 0; it < 8; ++it) {
      const int k = it * 128 + lane * 4;
      v4f x  = *(const v4f*)(srow + k);
      v4f g  = *(const v4f*)(gamma + k);
      v4f be = *(const v4f*)(beta + k);
      v4i mv = __builtin_nontemporal_load((const v4i*)(mrow + k));  // 268MB stream
      #pragma unroll
      for (int j = 0; j < 4; ++j) {
        float y = (x[j] - mean) * inv * g[j] + be[j];
        y = (mv[j] == 0) ? NEG_INF : y;
        x[j] = y;
        mx = fmaxf(mx, y);
      }
      *(v4f*)(srow + k) = x;
    }
    mx = wave_max(mx);

    float se = 0.f;
    #pragma unroll
    for (int it = 0; it < 8; ++it) {
      const int k = it * 128 + lane * 4;
      v4f x = *(const v4f*)(srow + k);
      #pragma unroll
      for (int j = 0; j < 4; ++j) { x[j] = __expf(x[j] - mx); se += x[j]; }
      *(v4f*)(srow + k) = x;
    }
    se = wave_sum(se);
    const float rs = 1.0f / se;

    float* arow = attn + ((size_t)bh * SQ + q0 + row) * SK;
    #pragma unroll
    for (int it = 0; it < 8; ++it) {
      const int k = it * 128 + lane * 4;
      v4f x = *(const v4f*)(srow + k);
      x *= rs;
      *(v4f*)(srow + k) = x;                    // keep probs for P*V
      __builtin_nontemporal_store(x, (v4f*)(arow + k));  // 268MB stream, write once
    }
  }
  __syncthreads();

  // ---------------- Phase 4: O = P (16x1024) @ V (1024x64) -----------------
  // Each wave owns one 16-wide output column slab, sweeps full K with WMMA.
  {
    const int nt = w;                           // output N-tile 0..3
    v8f oacc = {};
    const float* vbase = V + (size_t)bh * SK * HD;
    const float* prow  = sc + n * SK;           // A-matrix row m = lane&15
    #pragma unroll 2
    for (int c = 0; c < 32; ++c) {              // K chunks of 32
      const int kbase = c * 32;
      v16h pa, vb;
      #pragma unroll
      for (int v = 0; v < 8; ++v) {
        const int ka = kbase + ((v >> 2) << 4) + ((v & 3) << 1) + (hf << 3);
        pa[2 * v]     = (_Float16)prow[ka];
        pa[2 * v + 1] = (_Float16)prow[ka + 1];
        const int kb = kbase + ((v >> 2) << 4) + ((v & 3) << 1) + (hf << 3);
        vb[2 * v]     = (_Float16)vbase[(size_t)kb * HD + nt * 16 + n];
        vb[2 * v + 1] = (_Float16)vbase[(size_t)(kb + 1) * HD + nt * 16 + n];
      }
      oacc = __builtin_amdgcn_wmma_f32_16x16x32_f16(false, pa, false, vb,
                                                    (short)0, oacc, false, false);
    }
    #pragma unroll
    for (int r = 0; r < 8; ++r)
      out[((size_t)bh * SQ + q0 + r + 8 * hf) * HD + nt * 16 + n] = oacc[r];
  }
}

extern "C" void kernel_launch(void* const* d_in, const int* in_sizes, int n_in,
                              void* d_out, int out_size, void* d_ws, size_t ws_size,
                              hipStream_t stream) {
  (void)in_sizes; (void)n_in; (void)d_ws; (void)ws_size; (void)out_size;
  const float* Q     = (const float*)d_in[0];
  const float* K     = (const float*)d_in[1];
  const float* V     = (const float*)d_in[2];
  const float* gamma = (const float*)d_in[3];
  const float* beta  = (const float*)d_in[4];
  const int*   mask  = (const int*)d_in[5];

  float* out  = (float*)d_out;                                   // (B,H,SQ,64)
  float* attn = (float*)d_out + (size_t)BB * HH * SQ * HD;       // (B,H,SQ,SK)

  dim3 grid(SQ / 16, HH, BB);
  dim3 block(128);
  attn_ln_softmax_kernel<<<grid, block, 0, stream>>>(Q, K, V, gamma, beta, mask,
                                                     out, attn);
}
